// MyLayer_21930103013740
// MI455X (gfx1250) — compile-verified, matrix operand order checked by reading, not run on hardware
//
#include <hip/hip_runtime.h>
#include <hip/hip_bf16.h>
#include <stdint.h>

typedef __attribute__((ext_vector_type(16))) __bf16 v16bf;
typedef __attribute__((ext_vector_type(8)))  float  v8f;

#define DDIM 256   // feature dim
#define FDIM 16
#define ODIM 30
#define NTILE 16   // 256/16 column tiles
#define KCH   8    // 256/32 K chunks

// ---------------------------------------------------------------------------
// Kernel 1: W[i,j] = (j>i) ? dot(kernel[i,field[j],:], kernel[j,field[i],:]) : 0
// 256 blocks x 256 threads, one thread per element. f32 into workspace.
// ---------------------------------------------------------------------------
__global__ __launch_bounds__(256) void build_w_kernel(
    const float* __restrict__ kern, const int* __restrict__ field,
    float* __restrict__ W)
{
    const int i = blockIdx.x;
    const int j = threadIdx.x;
    float s = 0.0f;
    if (j > i) {
        const float* a = kern + (size_t)(i * FDIM + field[j]) * ODIM;
        const float* b = kern + (size_t)(j * FDIM + field[i]) * ODIM;
#pragma unroll
        for (int k = 0; k < ODIM; ++k) s += a[k] * b[k];
    }
    W[i * DDIM + j] = s;
}

// ---------------------------------------------------------------------------
// Kernel 2: out[b] = x_b^T * Wtri * x_b  via bf16 WMMA, f32 accumulate.
//  - W pre-swizzled into LDS in B-fragment layout (packed bf16 pairs, 128 KB)
//  - each wave: two 16-row tiles (32 rows) per pass; B fragments reused 2x
//  - reduction y.*x folded per 16-column tile (y never materialized)
// ---------------------------------------------------------------------------
__global__ __launch_bounds__(256) void qform_kernel(
    const float* __restrict__ x, const float* __restrict__ W,
    float* __restrict__ out, int Btotal)
{
    extern __shared__ uint32_t ldsW[];   // [NTILE][KCH][32 lanes][8 dwords] = 128 KB

    const int tid  = threadIdx.x;
    const int wave = tid >> 5;
    const int lane = tid & 31;
    const int g    = lane >> 4;   // half-wave group (K split of A/B layout)
    const int ln   = lane & 15;

    // ---- stage W into LDS, pre-swizzled into the 16-bit B-matrix layout ----
    // B fragment (n,k): lane L holds col N = 16n + (L&15); dword d holds the
    // packed bf16 pair for K = 32k + (d<4 ? 8g+2d : 16+8g+2(d-4)) , +1.
    for (int idx = tid; idx < NTILE * KCH * 32 * 8; idx += 256) {
        const int d  = idx & 7;
        const int L  = (idx >> 3) & 31;
        const int k  = (idx >> 8) & 7;
        const int n  = idx >> 11;
        const int gg = L >> 4;
        const int col = (n << 4) | (L & 15);
        const int kk  = (d < 4) ? (8 * gg + 2 * d) : (16 + 8 * gg + 2 * (d - 4));
        const int Kg  = (k << 5) + kk;
        const float w0 = W[(size_t)Kg * DDIM + col];
        const float w1 = W[(size_t)(Kg + 1) * DDIM + col];
        union { __bf16 h[2]; uint32_t u; } cv;
        cv.h[0] = (__bf16)w0;
        cv.h[1] = (__bf16)w1;
        ldsW[idx] = cv.u;
    }
    __syncthreads();

    // ---- main loop: each block pass covers 256 rows (8 waves x 32 rows) ----
    for (int chunk = blockIdx.x; chunk * 256 < Btotal; chunk += gridDim.x) {
        const int r0 = chunk * 256 + wave * 32;   // this wave's 32 rows

        // prefetch next chunk's slice of x (emits global_prefetch_b8)
        {
            const int nc = chunk + gridDim.x;
            if (nc * 256 < Btotal) {
                const float* pf = x + (size_t)(nc * 256 + wave * 32) * DDIM + lane * 8;
                __builtin_prefetch(pf, 0, 0);
            }
        }

        // Load A fragments for both row tiles, all 8 K-chunks, f32 -> bf16.
        // A layout (16-bit 16x32): lane L holds row M = L&15;
        //   g=0 lanes: K = {0..7, 16..23};  g=1 lanes: K = {8..15, 24..31}.
        v16bf afrag[2][KCH];
#pragma unroll
        for (int t = 0; t < 2; ++t) {
            const size_t row = (size_t)(r0 + t * 16 + ln);
            const float* xr = x + row * DDIM;
#pragma unroll
            for (int k = 0; k < KCH; ++k) {
                const float* p = xr + 32 * k + 8 * g;
                float4 f0 = *(const float4*)(p);
                float4 f1 = *(const float4*)(p + 4);
                float4 f2 = *(const float4*)(p + 16);
                float4 f3 = *(const float4*)(p + 20);
                v16bf a;
                a[0]=(__bf16)f0.x;  a[1]=(__bf16)f0.y;  a[2]=(__bf16)f0.z;  a[3]=(__bf16)f0.w;
                a[4]=(__bf16)f1.x;  a[5]=(__bf16)f1.y;  a[6]=(__bf16)f1.z;  a[7]=(__bf16)f1.w;
                a[8]=(__bf16)f2.x;  a[9]=(__bf16)f2.y;  a[10]=(__bf16)f2.z; a[11]=(__bf16)f2.w;
                a[12]=(__bf16)f3.x; a[13]=(__bf16)f3.y; a[14]=(__bf16)f3.z; a[15]=(__bf16)f3.w;
                afrag[t][k] = a;
            }
        }

        float acc[2][8];
#pragma unroll
        for (int t = 0; t < 2; ++t)
#pragma unroll
            for (int v = 0; v < 8; ++v) acc[t][v] = 0.0f;

        // Column-tile loop: accumulate C over K, then fold y.*x immediately.
#pragma unroll 1
        for (int n = 0; n < NTILE; ++n) {
            v8f c0 = {};
            v8f c1 = {};
            const uint32_t* bbase = ldsW + ((size_t)(n * KCH) * 32 + lane) * 8;
#pragma unroll
            for (int k = 0; k < KCH; ++k) {
                v16bf bf = *(const v16bf*)(bbase + (size_t)k * 32 * 8); // 32B contiguous
                c0 = __builtin_amdgcn_wmma_f32_16x16x32_bf16(
                        false, afrag[0][k], false, bf, (short)0, c0, false, false);
                c1 = __builtin_amdgcn_wmma_f32_16x16x32_bf16(
                        false, afrag[1][k], false, bf, (short)0, c1, false, false);
            }
            // C layout: VGPR v -> row (v + 8g), col N = 16n + ln
            const float* xc0 = x + (size_t)(r0 + 8 * g) * DDIM + 16 * n + ln;
            const float* xc1 = x + (size_t)(r0 + 16 + 8 * g) * DDIM + 16 * n + ln;
#pragma unroll
            for (int v = 0; v < 8; ++v) {
                acc[0][v] += c0[v] * xc0[(size_t)v * DDIM];
                acc[1][v] += c1[v] * xc1[(size_t)v * DDIM];
            }
        }

        // Cross-lane reduction over the 16 column lanes of each half-wave.
#pragma unroll
        for (int t = 0; t < 2; ++t) {
#pragma unroll
            for (int v = 0; v < 8; ++v) {
                float s = acc[t][v];
                s += __shfl_xor(s, 1, 32);
                s += __shfl_xor(s, 2, 32);
                s += __shfl_xor(s, 4, 32);
                s += __shfl_xor(s, 8, 32);
                if (ln == 0) out[r0 + t * 16 + v + 8 * g] = s;
            }
        }
    }
}

// ---------------------------------------------------------------------------
extern "C" void kernel_launch(void* const* d_in, const int* in_sizes, int n_in,
                              void* d_out, int out_size, void* d_ws, size_t ws_size,
                              hipStream_t stream) {
    const float* x     = (const float*)d_in[0];   // (B, 256) f32
    const float* kern  = (const float*)d_in[1];   // (256, 16, 30) f32
    const int*   field = (const int*)d_in[2];     // (256,) i32
    float* W   = (float*)d_ws;                    // 256*256 f32 = 256 KB scratch
    float* out = (float*)d_out;                   // (B,) f32
    const int Btotal = in_sizes[0] / DDIM;

    build_w_kernel<<<DDIM, DDIM, 0, stream>>>(kern, field, W);
    qform_kernel<<<128, 256, NTILE * KCH * 32 * 8 * sizeof(uint32_t), stream>>>(
        x, W, out, Btotal);
}
